// GraphAttentionLayer_37907381354912
// MI455X (gfx1250) — matrix operand order repaired and checked by reading, not compiled, not run on hardware
//
#include <hip/hip_runtime.h>

typedef __attribute__((ext_vector_type(2)))  float    v2f;
typedef __attribute__((ext_vector_type(8)))  float    v8f;
typedef __attribute__((ext_vector_type(16))) _Float16 v16h;

#define BATCH 8
#define NN    2048
#define FIN   256
#define FOUT  128
#define NEG_BIG (-9000000000000000.0f)
#define ALPHA 0.2f

// ---------------------------------------------------------------------------
// Kernel 1: Wh[b] = x[b] (2048x256) @ W^T (256x128), full-precision f32 WMMA.
// Also emits a transposed f16 copy WhT[b, f, j] for the aggregation GEMM.
// One wave per 16x16 output tile; V_WMMA_F32_16X16X4_F32, 64 K-steps.
// A-layout (32-bit 16x4): lane = m + 16*(k>=2), vgpr = k%2 -> contiguous pair.
// ---------------------------------------------------------------------------
__global__ __launch_bounds__(32)
void k_gemm_wh(const float* __restrict__ x, const float* __restrict__ W,
               float* __restrict__ Wh, _Float16* __restrict__ WhT) {
    const int lane = threadIdx.x;
    const int b  = blockIdx.z;
    const int i0 = blockIdx.x * 16;
    const int f0 = blockIdx.y * 16;
    const int hi = lane >> 4;     // 0: K pair {0,1}, 1: K pair {2,3}
    const int lo = lane & 15;     // A: row m; B: col n

    const float* xrow = x + ((size_t)b * NN + (i0 + lo)) * FIN;
    const float* wrow = W + (size_t)(f0 + lo) * FIN;

    v8f acc = {};
    for (int k0 = 0; k0 < FIN; k0 += 4) {
        const int kk = k0 + 2 * hi;
        v2f av; av.x = xrow[kk]; av.y = xrow[kk + 1];
        v2f bv; bv.x = wrow[kk]; bv.y = wrow[kk + 1];
        acc = __builtin_amdgcn_wmma_f32_16x16x4_f32(
            false, av, false, bv, (short)0, acc, false, false);
    }
    // C layout: lane lo = col, rows = v + 8*hi
#pragma unroll
    for (int v = 0; v < 8; ++v) {
        const int row = v + 8 * hi;
        const float val = acc[v];
        Wh [((size_t)b * NN   + (i0 + row)) * FOUT + f0 + lo] = val;
        WhT[((size_t)b * FOUT + (f0 + lo )) * NN   + i0 + row] = (_Float16)val;
    }
}

// ---------------------------------------------------------------------------
// Kernel 2: f_src = Wh @ a1, f_dst = Wh @ a2.  One wave per row, shfl reduce.
// ---------------------------------------------------------------------------
__global__ __launch_bounds__(128)
void k_fsrc_fdst(const float* __restrict__ Wh, const float* __restrict__ a,
                 float* __restrict__ fsrc, float* __restrict__ fdst) {
    const int lane = threadIdx.x & 31;
    const int r    = blockIdx.x * 4 + (threadIdx.x >> 5);   // [0, 16384)
    const float* row = Wh + (size_t)r * FOUT;
    float s = 0.f, d = 0.f;
#pragma unroll
    for (int t = 0; t < 4; ++t) {
        const float w = row[lane + 32 * t];
        s += w * a[lane + 32 * t];
        d += w * a[FOUT + lane + 32 * t];
    }
#pragma unroll
    for (int off = 16; off > 0; off >>= 1) {
        s += __shfl_xor(s, off, 32);
        d += __shfl_xor(d, off, 32);
    }
    if (lane == 0) { fsrc[r] = s; fdst[r] = d; }
}

// ---------------------------------------------------------------------------
// Kernel 3 (fused): per-row softmax stats + h' = softmax(e) @ Wh.
//
// 4 waves/block; each wave owns a 16-row tile x all 128 output columns.
// Pass 1: stream the wave's 16 adj rows ONCE (coalesced b32, 8 KB/row),
//         online max/sum-exp per row, and pack adjacency into a bitmask in
//         LDS via wave32 ballot (16 rows x 64 words = 4 KB per wave).
// Pass 2: rebuild P tiles from registers + 1 ds_load mask word per K-step,
//         f16 WMMA 16x16x32 against the contiguous WhT[b,f,j] B-operand.
// A layout per lane: rows lo, 16 consecutive K [16*hi..+15]
//   -> bits (16*hi+u) and (16*hi+8+u) of one 32-bit mask word.
// B layout per lane: col lo, 16 consecutive K -> one 32B contiguous load.
// ---------------------------------------------------------------------------
__global__ __launch_bounds__(128)
void k_attn_fused(const int* __restrict__ adj, const float* __restrict__ fsrc,
                  const float* __restrict__ fdst, const _Float16* __restrict__ WhT,
                  float* __restrict__ out) {
    __shared__ float        lds_d[NN];          //  8 KB: f_dst for this batch
    __shared__ unsigned int lds_m[4][16 * 64];  // 16 KB: adj bitmask per wave

    const int tid  = threadIdx.x;
    const int lane = tid & 31;
    const int w    = tid >> 5;
    const int t    = blockIdx.x * 4 + w;   // tile id, [0, 1024); 128 tiles/batch
    const int b    = t >> 7;               // all 4 waves share one batch (128%4==0)
    const int i0   = (t & 127) * 16;
    const int lo   = lane & 15;
    const int hi   = lane >> 4;

    for (int q = tid; q < NN; q += 128) lds_d[q] = fdst[(size_t)b * NN + q];
    __syncthreads();

    // ---- Pass 1: single streaming read of adj; stats + bitmask ----
    float mi = NEG_BIG, izi = 0.f;         // captured for row `lo` of this lane
    for (int rr = 0; rr < 16; ++rr) {
        const int*  arow = adj + ((size_t)b * NN + (i0 + rr)) * NN;
        const float si_r = fsrc[b * NN + i0 + rr];
        float m = NEG_BIG, Z = 0.f;
        for (int tt = 0; tt < 64; ++tt) {
            const int j  = tt * 32 + lane;
            const int aj = arow[j];
            const unsigned int bal = __builtin_amdgcn_ballot_w32(aj > 0);
            if (lane == 0) lds_m[w][rr * 64 + tt] = bal;
            float v = si_r + lds_d[j];
            v = v > 0.f ? v : ALPHA * v;
            v = aj > 0 ? v : NEG_BIG;
            const float nm = fmaxf(m, v);
            Z = Z * __expf(m - nm) + __expf(v - nm);
            m = nm;
        }
#pragma unroll
        for (int off = 16; off > 0; off >>= 1) {
            const float om = __shfl_xor(m, off, 32);
            const float oZ = __shfl_xor(Z, off, 32);
            const float nm = fmaxf(m, om);
            Z = Z * __expf(m - nm) + oZ * __expf(om - nm);
            m = nm;
        }
        if (rr == lo) { mi = m; izi = 1.0f / Z; }
    }

    // ---- Pass 2: WMMA aggregation, adj mask from LDS bits ----
    const float     si    = fsrc[b * NN + i0 + lo];
    const _Float16* bbase = WhT + (size_t)b * FOUT * NN;

    v8f acc[8] = {};
    for (int j0 = 0; j0 < NN; j0 += 32) {
        const unsigned int mw = lds_m[w][lo * 64 + (j0 >> 5)];
        v16h av;
#pragma unroll
        for (int u = 0; u < 8; ++u) {
            float v0 = si + lds_d[j0 + 16 * hi + u];
            v0 = v0 > 0.f ? v0 : ALPHA * v0;
            v0 = ((mw >> (16 * hi + u)) & 1u) ? v0 : NEG_BIG;
            av[u] = (_Float16)(__expf(v0 - mi) * izi);

            float v1 = si + lds_d[j0 + 16 * hi + 8 + u];
            v1 = v1 > 0.f ? v1 : ALPHA * v1;
            v1 = ((mw >> (16 * hi + 8 + u)) & 1u) ? v1 : NEG_BIG;
            av[8 + u] = (_Float16)(__expf(v1 - mi) * izi);
        }
#pragma unroll
        for (int ft = 0; ft < 8; ++ft) {
            const _Float16* bp = bbase + (size_t)(ft * 16 + lo) * NN + j0 + 16 * hi;
            const v16h bv = *(const v16h*)bp;   // 32B-aligned contiguous load
            acc[ft] = __builtin_amdgcn_wmma_f32_16x16x32_f16(
                false, av, false, bv, (short)0, acc[ft], false, false);
        }
    }
#pragma unroll
    for (int ft = 0; ft < 8; ++ft)
#pragma unroll
        for (int v = 0; v < 8; ++v)
            out[((size_t)b * NN + (i0 + v + 8 * hi)) * FOUT + ft * 16 + lo] = acc[ft][v];
}

// ---------------------------------------------------------------------------
extern "C" void kernel_launch(void* const* d_in, const int* in_sizes, int n_in,
                              void* d_out, int out_size, void* d_ws, size_t ws_size,
                              hipStream_t stream) {
    const float* x   = (const float*)d_in[0];
    const int*   adj = (const int*)d_in[1];
    const float* W   = (const float*)d_in[2];
    const float* a   = (const float*)d_in[3];
    float* out = (float*)d_out;

    char* ws = (char*)d_ws;
    float*    Wh   = (float*)(ws);                 //  8 MB  (8x2048x128 f32)
    _Float16* WhT  = (_Float16*)(ws + 8388608);    //  4 MB  (8x128x2048 f16)
    float*    fsrc = (float*)(ws + 12582912);      // 64 KB
    float*    fdst = (float*)(ws + 12648448);      // 64 KB

    k_gemm_wh   <<<dim3(NN / 16, FOUT / 16, BATCH), 32,  0, stream>>>(x, W, Wh, WhT);
    k_fsrc_fdst <<<dim3(BATCH * NN / 4),            128, 0, stream>>>(Wh, a, fsrc, fdst);
    k_attn_fused<<<dim3(BATCH * NN / 16 / 4),       128, 0, stream>>>(adj, fsrc, fdst, WhT, out);
}
